// recEdgeGNN_86500641341503
// MI455X (gfx1250) — compile-verified, matrix operand order checked by reading, not run on hardware
//
#include <hip/hip_runtime.h>

#define NNODES 50000
#define NEDGES 800000
// NFEAT = 128, 2H = 128, H = 64, NOUT = 128

typedef __attribute__((ext_vector_type(16))) __bf16 v16bf;
typedef __attribute__((ext_vector_type(8)))  float  v8f;

__device__ __forceinline__ float relu_f(float v) { return v > 0.f ? v : 0.f; }

// ---------------------------------------------------------------------------
// WMMA fragment layouts (ISA 7.12.2, wave32):
//  A (16x32 bf16): lane L<16 -> row M=L, K in {0..7, 16..23};
//                  lane L+16 -> row M=L, K in {8..15, 24..31}.
//  Per lane the two 8-element K-chunks are contiguous -> 2x b128 LDS loads.
//  B (32x16 KxN): same K distribution per lane-half, N = lane&15.
//  C/D (16x16 f32): element j: M = j + 8*(lane>=16), N = lane&15.
// ---------------------------------------------------------------------------

__device__ __forceinline__ v16bf load_A_frag(const __bf16* As, int strideA, int kbase, int lane) {
  v16bf a;
  int r  = lane & 15;
  int kh = (lane >> 4) & 1;
#pragma unroll
  for (int i = 0; i < 8; ++i) {
    int k0 = ((i >> 2) << 4) + (kh << 3) + ((i & 3) << 1);
    a[2 * i]     = As[r * strideA + kbase + k0];
    a[2 * i + 1] = As[r * strideA + kbase + k0 + 1];
  }
  return a;
}

// B pre-packed fragment-major: P[((kt*ntiles + nt)*32 + lane)*16 + i] holds the
// 16 bf16 values lane needs for tile (kt,nt). One 32B contiguous load per lane.
__device__ __forceinline__ v16bf load_B_packed(const __bf16* P, int ntiles, int kt, int nt, int lane) {
  const v16bf* p = (const v16bf*)(P + (((kt * ntiles + nt) * 32 + lane) << 4));
  return *p;
}

__device__ __forceinline__ v8f wmma_bf16(v16bf a, v16bf b, v8f c) {
  return __builtin_amdgcn_wmma_f32_16x16x32_bf16(false, a, false, b, (short)0, c, false, false);
}

// ---------------- pack f32 row-major [K,Ncols] weight -> bf16 fragment-major ----------------
__global__ void k_packB(const float* __restrict__ W, __bf16* __restrict__ P, int K, int Ncols) {
  int total = K * Ncols;
  int ntiles = Ncols >> 4;
  for (int idx = blockIdx.x * blockDim.x + threadIdx.x; idx < total; idx += gridDim.x * blockDim.x) {
    int i2   = idx & 15;
    int lane = (idx >> 4) & 31;
    int t    = idx >> 9;            // fragment tile index
    int kt = t / ntiles, nt = t - kt * ntiles;
    int i = i2 >> 1, pair = i2 & 1;
    int k = kt * 32 + ((i >> 2) << 4) + ((lane >> 4) << 3) + ((i & 3) << 1) + pair;
    int n = nt * 16 + (lane & 15);
    P[idx] = (__bf16)W[k * Ncols + n];
  }
}

__global__ void k_zero_f32(float* __restrict__ p, int n) {
  for (int i = blockIdx.x * blockDim.x + threadIdx.x; i < n; i += gridDim.x * blockDim.x)
    p[i] = 0.f;
}

__global__ void k_f32_to_bf16(const float* __restrict__ src, __bf16* __restrict__ dst, int n) {
  for (int i = blockIdx.x * blockDim.x + threadIdx.x; i < n; i += gridDim.x * blockDim.x)
    dst[i] = (__bf16)src[i];
}

// ---------------- h = relu(x @ W_n11 + b_n11)  [N,128]x[128,128] ----------------
// 256 threads = 8 waves; block = 16 rows x 128 cols; wave w -> col tile w.
__global__ void k_n11(const float* __restrict__ x, const __bf16* __restrict__ Wp,
                      const float* __restrict__ bias, __bf16* __restrict__ hbf) {
  __shared__ __bf16 As[16 * 128];
  int tid  = threadIdx.x;
  int row0 = blockIdx.x * 16;
#pragma unroll
  for (int i = 0; i < 8; ++i) {
    int l = tid * 8 + i;
    As[l] = (__bf16)x[(long)row0 * 128 + l];
  }
  __syncthreads();
  int wave = tid >> 5, lane = tid & 31;
  v8f c = {};
#pragma unroll
  for (int kt = 0; kt < 4; ++kt) {
    v16bf a = load_A_frag(As, 128, kt * 32, lane);
    v16bf b = load_B_packed(Wp, 8, kt, wave, lane);
    c = wmma_bf16(a, b, c);
  }
  int kh = lane >> 4, col = wave * 16 + (lane & 15);
  float bv = bias[col];
#pragma unroll
  for (int j = 0; j < 8; ++j) {
    int m = j + kh * 8;
    hbf[(long)(row0 + m) * 128 + col] = (__bf16)relu_f(c[j] + bv);
  }
}

// ---------------- fused node MLP: h2 = relu(relu(h@W21+b21)@W22+b22) ----------------
// 128 threads = 4 waves; 16 rows per block; output [N,64] bf16.
__global__ void k_node2(const __bf16* __restrict__ hbf,
                        const __bf16* __restrict__ W21p, const float* __restrict__ b21,
                        const __bf16* __restrict__ W22p, const float* __restrict__ b22,
                        __bf16* __restrict__ h2bf) {
  __shared__ __bf16 As[16 * 128];
  __shared__ __bf16 Ts[16 * 64];
  int tid = threadIdx.x;
  int row0 = blockIdx.x * 16;
  const unsigned int* g = (const unsigned int*)(hbf + (long)row0 * 128);
  unsigned int* l = (unsigned int*)As;
#pragma unroll
  for (int i = 0; i < 8; ++i) l[tid + i * 128] = g[tid + i * 128];
  __syncthreads();

  int wave = tid >> 5, lane = tid & 31;
  v8f c = {};
#pragma unroll
  for (int kt = 0; kt < 4; ++kt) {
    v16bf a = load_A_frag(As, 128, kt * 32, lane);
    v16bf b = load_B_packed(W21p, 4, kt, wave, lane);
    c = wmma_bf16(a, b, c);
  }
  int kh = lane >> 4, col = wave * 16 + (lane & 15);
  float bv = b21[col];
#pragma unroll
  for (int j = 0; j < 8; ++j) {
    int m = j + kh * 8;
    Ts[m * 64 + col] = (__bf16)relu_f(c[j] + bv);
  }
  __syncthreads();

  v8f c2 = {};
#pragma unroll
  for (int kt = 0; kt < 2; ++kt) {
    v16bf a = load_A_frag(Ts, 64, kt * 32, lane);
    v16bf b = load_B_packed(W22p, 4, kt, wave, lane);
    c2 = wmma_bf16(a, b, c2);
  }
  bv = b22[col];
#pragma unroll
  for (int j = 0; j < 8; ++j) {
    int m = j + kh * 8;
    h2bf[(long)(row0 + m) * 64 + col] = (__bf16)relu_f(c2[j] + bv);
  }
}

// ---------------- edge pipeline: gather -> 2 GEMMs -> scatter-add ----------------
// 128 threads = 4 waves; 16 edges per block.
__global__ void k_edge(const __bf16* __restrict__ h2bf,
                       const int* __restrict__ src, const int* __restrict__ dst,
                       const __bf16* __restrict__ We11p, const float* __restrict__ be11,
                       const __bf16* __restrict__ We12p, const float* __restrict__ be12,
                       float* __restrict__ acc) {
  __shared__ __bf16 As[16 * 128];
  __shared__ __bf16 Ts[16 * 64];
  __shared__ int sidx[16], didx[16];
  int tid = threadIdx.x;
  int e0 = blockIdx.x * 16;
  if (tid < 16) {
    sidx[tid] = src[e0 + tid];
    didx[tid] = dst[e0 + tid];
  }
  if (e0 + 16 < NEDGES && tid == 16) {
    __builtin_prefetch(&src[e0 + 16], 0, 1);
    __builtin_prefetch(&dst[e0 + 16], 0, 1);
  }
  __syncthreads();

  // gather: row r of A = concat(h2[src[r]], h2[dst[r]]); each thread 16 bf16 (32B)
  {
    int r = tid >> 3, part = tid & 7;
    int node = (part < 4) ? sidx[r] : didx[r];
    int cbase = (part & 3) * 16;
    const unsigned int* g = (const unsigned int*)(h2bf + (long)node * 64 + cbase);
    unsigned int* l = (unsigned int*)(As + r * 128 + part * 16);
#pragma unroll
    for (int i = 0; i < 8; ++i) l[i] = g[i];
  }
  __syncthreads();

  int wave = tid >> 5, lane = tid & 31;
  v8f c = {};
#pragma unroll
  for (int kt = 0; kt < 4; ++kt) {
    v16bf a = load_A_frag(As, 128, kt * 32, lane);
    v16bf b = load_B_packed(We11p, 4, kt, wave, lane);
    c = wmma_bf16(a, b, c);
  }
  int kh = lane >> 4, col = wave * 16 + (lane & 15);
  float bv = be11[col];
#pragma unroll
  for (int j = 0; j < 8; ++j) {
    int m = j + kh * 8;
    Ts[m * 64 + col] = (__bf16)relu_f(c[j] + bv);
  }
  __syncthreads();

  v8f c2 = {};
#pragma unroll
  for (int kt = 0; kt < 2; ++kt) {
    v16bf a = load_A_frag(Ts, 64, kt * 32, lane);
    v16bf b = load_B_packed(We12p, 4, kt, wave, lane);
    c2 = wmma_bf16(a, b, c2);
  }
  bv = be12[col];
#pragma unroll
  for (int j = 0; j < 8; ++j) {
    int m = j + kh * 8;
    float v = relu_f(c2[j] + bv);
    atomicAdd(&acc[(long)sidx[m] * 128 + col], v);       // segment_sum over src -> cols [0,64)
    atomicAdd(&acc[(long)didx[m] * 128 + 64 + col], v);  // segment_sum over dst -> cols [64,128)
  }
}

// ---------------- final reduction + pred ----------------
__global__ void k_colsum(const float* __restrict__ h, float* __restrict__ colsum) {
  int col = threadIdx.x;  // 128 threads
  float s = 0.f;
  for (int r = blockIdx.x; r < NNODES; r += gridDim.x) s += h[(long)r * 128 + col];
  atomicAdd(&colsum[col], s);
}

__global__ void k_pred(const float* __restrict__ colsum, const float* __restrict__ W,
                       const float* __restrict__ bias, float* __restrict__ pred) {
  int j = threadIdx.x;  // 128 threads
  float s = bias[j];
  const float inv = 1.0f / (float)NNODES;
#pragma unroll 8
  for (int k = 0; k < 128; ++k) s += colsum[k] * inv * W[k * 128 + j];
  pred[j] = s;
}

extern "C" void kernel_launch(void* const* d_in, const int* in_sizes, int n_in,
                              void* d_out, int out_size, void* d_ws, size_t ws_size,
                              hipStream_t stream) {
  const float* x    = (const float*)d_in[0];
  const float* Wn11 = (const float*)d_in[1];
  const float* bn11 = (const float*)d_in[2];
  const float* Wn21 = (const float*)d_in[3];
  const float* bn21 = (const float*)d_in[4];
  const float* Wn22 = (const float*)d_in[5];
  const float* bn22 = (const float*)d_in[6];
  const float* We11 = (const float*)d_in[7];
  const float* be11 = (const float*)d_in[8];
  const float* We12 = (const float*)d_in[9];
  const float* be12 = (const float*)d_in[10];
  const float* We2p = (const float*)d_in[11];
  const float* be2p = (const float*)d_in[12];
  const int*   srcI = (const int*)d_in[13];
  const int*   dstI = (const int*)d_in[14];

  // workspace layout (bf16 activations + packed bf16 weights + colsum)
  __bf16* hbf   = (__bf16*)d_ws;                 // N*128
  __bf16* h2bf  = hbf + (long)NNODES * 128;      // N*64
  __bf16* Wn11p = h2bf + (long)NNODES * 64;      // 128*128
  __bf16* Wn21p = Wn11p + 128 * 128;             // 128*64
  __bf16* Wn22p = Wn21p + 128 * 64;              // 64*64
  __bf16* We11p = Wn22p + 64 * 64;               // 128*64
  __bf16* We12p = We11p + 128 * 64;              // 64*64
  float*  colsum = (float*)(We12p + 64 * 64);    // 128

  float* acc  = (float*)d_out;                   // N*128 f32 accumulator / final h
  float* pred = acc + (long)NNODES * 128;        // 128 f32

  // 1) pack weights into bf16 WMMA fragment-major layout
  k_packB<<<64, 256, 0, stream>>>(Wn11, Wn11p, 128, 128);
  k_packB<<<32, 256, 0, stream>>>(Wn21, Wn21p, 128, 64);
  k_packB<<<16, 256, 0, stream>>>(Wn22, Wn22p, 64, 64);
  k_packB<<<32, 256, 0, stream>>>(We11, We11p, 128, 64);
  k_packB<<<16, 256, 0, stream>>>(We12, We12p, 64, 64);

  // 2) h = relu(x @ W_n11 + b)
  k_n11<<<NNODES / 16, 256, 0, stream>>>(x, Wn11p, bn11, hbf);

  // 3) message-passing iterations
  for (int it = 0; it < 3; ++it) {
    k_node2<<<NNODES / 16, 128, 0, stream>>>(hbf, Wn21p, bn21, Wn22p, bn22, h2bf);
    k_zero_f32<<<2048, 256, 0, stream>>>(acc, NNODES * 128);
    k_edge<<<NEDGES / 16, 128, 0, stream>>>(h2bf, srcI, dstI, We11p, be11, We12p, be12, acc);
    if (it < 2) k_f32_to_bf16<<<2048, 256, 0, stream>>>(acc, hbf, NNODES * 128);
  }

  // 4) pred = mean(h,0) @ W_e2p + b
  k_zero_f32<<<1, 128, 0, stream>>>(colsum, 128);
  k_colsum<<<128, 128, 0, stream>>>(acc, colsum);
  k_pred<<<1, 128, 0, stream>>>(colsum, We2p, be2p, pred);
}
